// CGRUCell_51677046505498
// MI455X (gfx1250) — matrix operand order repaired and checked by reading, not compiled, not run on hardware
//
#include <hip/hip_runtime.h>
#include <math.h>

#define UNITS   2048
#define B_ROWS  1024
#define HDIM    4096      // 2*UNITS
#define LDA     4096      // row stride of activations / ws matrices (f32)
#define LDB     6144      // row stride of weight matrices (f32)
#define BM      128
#define BN      128
#define BK      32
#define KSTEPS  (UNITS / BK)   // 64 slabs per 2048-K segment
#define LSTR    40        // LDS tile row stride in bf16 (padded, rows stay 16B-aligned)
#define NWORDS  ((size_t)B_ROWS * HDIM)

typedef __bf16 bf16;
typedef __attribute__((ext_vector_type(2)))  __bf16 v2bf;
typedef __attribute__((ext_vector_type(8)))  __bf16 v8bf;
typedef __attribute__((ext_vector_type(16))) __bf16 v16bf;
typedef __attribute__((ext_vector_type(8)))  float  v8f;

__device__ __forceinline__ float hsig(float x) {
    return fminf(fmaxf(0.2f * x + 0.5f, 0.0f), 1.0f);
}

// ---- WMMA compute over one staged BK=32 slab --------------------------------
// A in LDS: [m][k] row-major, stride LSTR. Lane (lh = lane>>4, ml = lane&15)
// holds row ml; 16 elems are K = lh*8 + {0..7} and lh*8+16+{0..7} -> 2x b128.
// B in LDS: transposed [n][k], stride LSTR. Lane holds col ml, K same as A.
__device__ __forceinline__ void wmma_slab(const bf16* __restrict__ Alds,
                                          const bf16* __restrict__ Blds,
                                          int m_off, int n_off, int lh, int ml,
                                          v8f acc[4][2]) {
    v16bf a[4], b[2];
#pragma unroll
    for (int fm = 0; fm < 4; ++fm) {
        const bf16* p = Alds + (m_off + fm * 16 + ml) * LSTR + lh * 8;
        v8bf lo = *reinterpret_cast<const v8bf*>(p);
        v8bf hi = *reinterpret_cast<const v8bf*>(p + 16);
        a[fm] = __builtin_shufflevector(lo, hi, 0,1,2,3,4,5,6,7,8,9,10,11,12,13,14,15);
    }
#pragma unroll
    for (int fn = 0; fn < 2; ++fn) {
        const bf16* p = Blds + (n_off + fn * 16 + ml) * LSTR + lh * 8;
        v8bf lo = *reinterpret_cast<const v8bf*>(p);
        v8bf hi = *reinterpret_cast<const v8bf*>(p + 16);
        b[fn] = __builtin_shufflevector(lo, hi, 0,1,2,3,4,5,6,7,8,9,10,11,12,13,14,15);
    }
#pragma unroll
    for (int fm = 0; fm < 4; ++fm)
#pragma unroll
        for (int fn = 0; fn < 2; ++fn)
            acc[fm][fn] = __builtin_amdgcn_wmma_f32_16x16x32_bf16(
                false, a[fm], false, b[fn], (short)0, acc[fm][fn], false, false);
}

// ---- register <-> global / LDS staging -------------------------------------
// B: thread covers a 2(K)x4(N) patch twice: reads two adjacent K rows (float4
// along N), packs vertical (k,k+1) bf16 pairs, stores v2bf into transposed tile.
__device__ __forceinline__ void loadB(float4 rb[2][2], const float* __restrict__ Bb,
                                      int k0, int tid) {
    const int cg = tid & 31, kp = tid >> 5;
#pragma unroll
    for (int i = 0; i < 2; ++i) {
        const int p2 = kp + 8 * i;                     // k-pair index 0..15
        const float* p = Bb + (size_t)(k0 + 2 * p2) * LDB + cg * 4;
        rb[i][0] = *reinterpret_cast<const float4*>(p);
        rb[i][1] = *reinterpret_cast<const float4*>(p + LDB);
    }
}

__device__ __forceinline__ void storeB(bf16* __restrict__ Blds,
                                       const float4 rb[2][2], float sgn, int tid) {
    const int cg = tid & 31, kp = tid >> 5;
#pragma unroll
    for (int i = 0; i < 2; ++i) {
        const int p2 = kp + 8 * i;
        const float* r0 = reinterpret_cast<const float*>(&rb[i][0]);
        const float* r1 = reinterpret_cast<const float*>(&rb[i][1]);
#pragma unroll
        for (int j = 0; j < 4; ++j) {
            v2bf w = {(bf16)(r0[j] * sgn), (bf16)(r1[j] * sgn)};
            *reinterpret_cast<v2bf*>(Blds + (cg * 4 + j) * LSTR + 2 * p2) = w;
        }
    }
}

// A: thread covers 8 contiguous K of one row, twice -> one b128 LDS store each.
__device__ __forceinline__ void loadA(float4 ra[2][2], const float* __restrict__ Ab,
                                      int row0, int k0, int tid) {
    const int q = tid & 3, rr = tid >> 2;
#pragma unroll
    for (int i = 0; i < 2; ++i) {
        const float* p = Ab + (size_t)(row0 + rr + 64 * i) * LDA + k0 + q * 8;
        ra[i][0] = *reinterpret_cast<const float4*>(p);
        ra[i][1] = *reinterpret_cast<const float4*>(p + 4);
    }
}

__device__ __forceinline__ void storeA(bf16* __restrict__ Alds,
                                       const float4 ra[2][2], int tid) {
    const int q = tid & 3, rr = tid >> 2;
#pragma unroll
    for (int i = 0; i < 2; ++i) {
        const float* f0 = reinterpret_cast<const float*>(&ra[i][0]);
        const float* f1 = reinterpret_cast<const float*>(&ra[i][1]);
        v8bf w = {(bf16)f0[0], (bf16)f0[1], (bf16)f0[2], (bf16)f0[3],
                  (bf16)f1[0], (bf16)f1[1], (bf16)f1[2], (bf16)f1[3]};
        *reinterpret_cast<v8bf*>(Alds + (rr + 64 * i) * LSTR + q * 8) = w;
    }
}

// Phase-2 A operand computed on the fly: hard_sigmoid(Rpre) * h
__device__ __forceinline__ void loadA_rh(float4 rra[2][2], float4 rha[2][2],
                                         const float* __restrict__ Rp,
                                         const float* __restrict__ Hp,
                                         int row0, int k0, int tid) {
    const int q = tid & 3, rr = tid >> 2;
#pragma unroll
    for (int i = 0; i < 2; ++i) {
        const size_t off = (size_t)(row0 + rr + 64 * i) * LDA + k0 + q * 8;
        rra[i][0] = *reinterpret_cast<const float4*>(Rp + off);
        rra[i][1] = *reinterpret_cast<const float4*>(Rp + off + 4);
        rha[i][0] = *reinterpret_cast<const float4*>(Hp + off);
        rha[i][1] = *reinterpret_cast<const float4*>(Hp + off + 4);
    }
}

__device__ __forceinline__ void storeA_rh(bf16* __restrict__ Alds,
                                          const float4 rra[2][2],
                                          const float4 rha[2][2], int tid) {
    const int q = tid & 3, rr = tid >> 2;
#pragma unroll
    for (int i = 0; i < 2; ++i) {
        const float* r0 = reinterpret_cast<const float*>(&rra[i][0]);
        const float* r1 = reinterpret_cast<const float*>(&rra[i][1]);
        const float* h0 = reinterpret_cast<const float*>(&rha[i][0]);
        const float* h1 = reinterpret_cast<const float*>(&rha[i][1]);
        v8bf w = {(bf16)(hsig(r0[0]) * h0[0]), (bf16)(hsig(r0[1]) * h0[1]),
                  (bf16)(hsig(r0[2]) * h0[2]), (bf16)(hsig(r0[3]) * h0[3]),
                  (bf16)(hsig(r1[0]) * h1[0]), (bf16)(hsig(r1[1]) * h1[1]),
                  (bf16)(hsig(r1[2]) * h1[2]), (bf16)(hsig(r1[3]) * h1[3])};
        *reinterpret_cast<v8bf*>(Alds + (rr + 64 * i) * LSTR + q * 8) = w;
    }
}

// ---- phase 1: Zpre, Rpre (K=8192) and Xh (K=4096) pre-activations ----------
__global__ __launch_bounds__(256) void cgru_gates(
    const float* __restrict__ inputs, const float* __restrict__ h,
    const float* __restrict__ rk,  const float* __restrict__ ik,
    const float* __restrict__ rrk, const float* __restrict__ irk,
    const float* __restrict__ rb,  const float* __restrict__ ib,
    float* __restrict__ ws)
{
    __shared__ bf16 Alds[BM * LSTR];
    __shared__ bf16 Blds[BN * LSTR];

    const int tid   = threadIdx.x;
    const int row0  = blockIdx.y * BM;
    const int colg  = blockIdx.x * BN;          // 0..12287 (3 gates x 4096)
    const int gate  = colg / HDIM;              // 0=z, 1=r, 2=h-candidate
    const int c     = colg - gate * HDIM;       // col within gate output
    const int im    = (c >= UNITS) ? 1 : 0;     // imaginary half?
    const int n0    = c - im * UNITS;           // col within 2048 weight block

    const float* Wr = rk  + gate * UNITS + n0;
    const float* Wi = ik  + gate * UNITS + n0;
    const float* Rr = rrk + gate * UNITS + n0;
    const float* Ri = irk + gate * UNITS + n0;

    const float* Aseg[4]; const float* Bseg[4]; float sseg[4];
    if (!im) {  // real half:  x1@Wr + x2@Wi (+ h1@Rr + h2@Ri)
        Aseg[0] = inputs;         Bseg[0] = Wr; sseg[0] =  1.f;
        Aseg[1] = inputs + UNITS; Bseg[1] = Wi; sseg[1] =  1.f;
        Aseg[2] = h;              Bseg[2] = Rr; sseg[2] =  1.f;
        Aseg[3] = h + UNITS;      Bseg[3] = Ri; sseg[3] =  1.f;
    } else {    // imag half: -x1@Wi + x2@Wr (- h1@Ri + h2@Rr)
        Aseg[0] = inputs;         Bseg[0] = Wi; sseg[0] = -1.f;
        Aseg[1] = inputs + UNITS; Bseg[1] = Wr; sseg[1] =  1.f;
        Aseg[2] = h;              Bseg[2] = Ri; sseg[2] = -1.f;
        Aseg[3] = h + UNITS;      Bseg[3] = Rr; sseg[3] =  1.f;
    }
    const int nseg  = (gate < 2) ? 4 : 2;
    const int iters = nseg * KSTEPS;

    const int wid = tid >> 5;
    const int m_off = (wid >> 2) * 64;
    const int n_off = (wid & 3) * 32;
    const int lane = tid & 31;
    const int lh = lane >> 4, ml = lane & 15;

    v8f acc[4][2] = {};
    float4 ra[2][2], rbuf[2][2];

    // software pipeline: regs hold slab `it`; next slab's globals issue
    // before the WMMA block so L2/HBM latency overlaps matrix work.
    const float* Ab = Aseg[0]; const float* Bb = Bseg[0]; float sg = sseg[0];
    loadA(ra, Ab, row0, 0, tid);
    loadB(rbuf, Bb, 0, tid);

    for (int it = 0; it < iters; ++it) {
        __syncthreads();                 // previous slab's WMMA reads done
        storeA(Alds, ra, tid);
        storeB(Blds, rbuf, sg, tid);
        __syncthreads();                 // tile visible to all waves
        const int nx = it + 1;
        if (nx < iters) {
            const int s = nx >> 6, k0 = (nx & 63) * BK;
            Ab = Aseg[s]; Bb = Bseg[s]; sg = sseg[s];
            loadA(ra, Ab, row0, k0, tid);
            loadB(rbuf, Bb, k0, tid);
        }
        wmma_slab(Alds, Blds, m_off, n_off, lh, ml, acc);
    }

    const float* biasp = (im ? ib : rb) + gate * UNITS + n0;
    float* outp = ws + (size_t)gate * NWORDS;
#pragma unroll
    for (int fm = 0; fm < 4; ++fm)
#pragma unroll
        for (int fn = 0; fn < 2; ++fn) {
            const int col = n_off + fn * 16 + ml;
            const float bv = biasp[col];
#pragma unroll
            for (int r = 0; r < 8; ++r) {
                const int rg = row0 + m_off + fm * 16 + lh * 8 + r;
                outp[(size_t)rg * HDIM + c + col] = acc[fm][fn][r] + bv;
            }
        }
}

// ---- phase 2: Y = (hsig(Rpre)*h) @ Rh, fused GRU combine -------------------
__global__ __launch_bounds__(256) void cgru_out(
    const float* __restrict__ h,
    const float* __restrict__ rrk, const float* __restrict__ irk,
    const float* __restrict__ ws, float* __restrict__ out)
{
    __shared__ bf16 Alds[BM * LSTR];
    __shared__ bf16 Blds[BN * LSTR];

    const int tid  = threadIdx.x;
    const int row0 = blockIdx.y * BM;
    const int c    = blockIdx.x * BN;          // 0..4095
    const int im   = (c >= UNITS) ? 1 : 0;
    const int n0   = c - im * UNITS;

    const float* Zpre = ws;
    const float* Rpre = ws + NWORDS;
    const float* Xh   = ws + 2 * NWORDS;

    const float* Rrh = rrk + 2 * UNITS + n0;   // candidate-gate weight slices
    const float* Rih = irk + 2 * UNITS + n0;

    const float* Rseg[2]; const float* Hseg[2]; const float* Bseg[2]; float sseg[2];
    if (!im) {  //  rh1@Rr + rh2@Ri
        Rseg[0] = Rpre;         Hseg[0] = h;         Bseg[0] = Rrh; sseg[0] =  1.f;
        Rseg[1] = Rpre + UNITS; Hseg[1] = h + UNITS; Bseg[1] = Rih; sseg[1] =  1.f;
    } else {    // -rh1@Ri + rh2@Rr
        Rseg[0] = Rpre;         Hseg[0] = h;         Bseg[0] = Rih; sseg[0] = -1.f;
        Rseg[1] = Rpre + UNITS; Hseg[1] = h + UNITS; Bseg[1] = Rrh; sseg[1] =  1.f;
    }
    const int iters = 2 * KSTEPS;

    const int wid = tid >> 5;
    const int m_off = (wid >> 2) * 64;
    const int n_off = (wid & 3) * 32;
    const int lane = tid & 31;
    const int lh = lane >> 4, ml = lane & 15;

    v8f acc[4][2] = {};
    float4 rra[2][2], rha[2][2], rbuf[2][2];

    const float* Rp = Rseg[0]; const float* Hp = Hseg[0];
    const float* Bb = Bseg[0]; float sg = sseg[0];
    loadA_rh(rra, rha, Rp, Hp, row0, 0, tid);
    loadB(rbuf, Bb, 0, tid);

    for (int it = 0; it < iters; ++it) {
        __syncthreads();
        storeA_rh(Alds, rra, rha, tid);
        storeB(Blds, rbuf, sg, tid);
        __syncthreads();
        const int nx = it + 1;
        if (nx < iters) {
            const int s = nx >> 6, k0 = (nx & 63) * BK;
            Rp = Rseg[s]; Hp = Hseg[s]; Bb = Bseg[s]; sg = sseg[s];
            loadA_rh(rra, rha, Rp, Hp, row0, k0, tid);
            loadB(rbuf, Bb, k0, tid);
        }
        wmma_slab(Alds, Blds, m_off, n_off, lh, ml, acc);
    }

    // h_out = z*h + (1-z)*tanh(Xh + Y)
#pragma unroll
    for (int fm = 0; fm < 4; ++fm)
#pragma unroll
        for (int fn = 0; fn < 2; ++fn) {
            const int col = n_off + fn * 16 + ml;
#pragma unroll
            for (int r = 0; r < 8; ++r) {
                const int rg = row0 + m_off + fm * 16 + lh * 8 + r;
                const size_t idx = (size_t)rg * HDIM + c + col;
                const float z = hsig(Zpre[idx]);
                const float hh = tanhf(Xh[idx] + acc[fm][fn][r]);
                out[idx] = z * h[idx] + (1.f - z) * hh;
            }
        }
}

extern "C" void kernel_launch(void* const* d_in, const int* in_sizes, int n_in,
                              void* d_out, int out_size, void* d_ws, size_t ws_size,
                              hipStream_t stream) {
    const float* inputs = (const float*)d_in[0];
    const float* h      = (const float*)d_in[1];
    const float* rk     = (const float*)d_in[2];
    const float* ik     = (const float*)d_in[3];
    const float* rrk    = (const float*)d_in[4];
    const float* irk    = (const float*)d_in[5];
    const float* rb     = (const float*)d_in[6];
    const float* ib     = (const float*)d_in[7];
    float* out = (float*)d_out;
    float* ws  = (float*)d_ws;   // needs 3 * 1024 * 4096 * 4B = 48 MB

    dim3 blk(256);
    dim3 g1(96, 8);   // 12288/128 N-tiles x 1024/128 M-tiles
    cgru_gates<<<g1, blk, 0, stream>>>(inputs, h, rk, ik, rrk, irk, rb, ib, ws);
    dim3 g2(32, 8);   // 4096/128 x 8
    cgru_out<<<g2, blk, 0, stream>>>(h, rrk, irk, ws, out);
}
